// AttMLR_25434796327681
// MI455X (gfx1250) — matrix operand order repaired and checked by reading, not compiled, not run on hardware
//
#include <hip/hip_runtime.h>
#include <math.h>

// ---------------------------------------------------------------------------
// MLR block-diagonal attention for MI455X (gfx1250), wave32 + WMMA bf16 + TDM.
//   B=1, T=2048, C=1024, H=16, D=64; ranks {32,16,16}, blocks {2048,1024,512}
// GEMM tiles are staged into LDS by the Tensor Data Mover (tensor_load_to_lds,
// synchronized with s_wait_tensorcnt); all matmuls run on
// v_wmma_f32_16x16x32_bf16 with fp32 accumulation.
// ---------------------------------------------------------------------------

typedef __attribute__((ext_vector_type(16))) __bf16       v16bf;
typedef __attribute__((ext_vector_type(8)))  float        v8f;
typedef __attribute__((ext_vector_type(4)))  unsigned int v4u;
typedef __attribute__((ext_vector_type(8)))  int          v8i;
typedef __attribute__((ext_vector_type(4)))  int          v4i;

#define T_SEQ  2048
#define C_DIM  1024
#define H_NUM  16
#define D_HEAD 64

// A-matrix (16x32, 16-bit) per-lane element -> K index (contiguous: k = 16*half + e, matches B)
__device__ __forceinline__ int kmapA(int e, int half) {
    return e + 16 * half;
}

__device__ __forceinline__ v8f wmma_bf16(v16bf a, v16bf b, v8f c) {
    // (neg_a, A, neg_b, B, c_mod, C, reuse_a, reuse_b)
    return __builtin_amdgcn_wmma_f32_16x16x32_bf16(false, a, false, b,
                                                   (short)0, c, false, false);
}

// ---------------------------------------------------------------------------
// TDM: async 2-D tile load (fp32 elements) global -> LDS.
// Descriptor per CDNA5 ISA §8.3/8.4: group0 = {count=1, lds_addr, global_addr,
// type=2}; group1 = {data_size=4B, tensor dims, tile dims, dim0 stride}.
// Tracked by TENSORcnt (s_wait_tensorcnt 0 before consuming the LDS tile).
// ---------------------------------------------------------------------------
__device__ __forceinline__ void tdm_load_tile_f32(float* lds_dst,
                                                  const float* gsrc,
                                                  unsigned stride_elems,
                                                  unsigned tile_w,
                                                  unsigned tile_h) {
    const unsigned long long ga = (unsigned long long)(size_t)gsrc;
    const unsigned lds_addr = (unsigned)(size_t)lds_dst;   // addr[31:0] = LDS offset
    v4u g0;
    g0[0] = 1u;                                            // count=1, user D#
    g0[1] = lds_addr;                                      // lds_addr[31:0]
    g0[2] = (unsigned)(ga & 0xFFFFFFFFull);                // global_addr lo
    g0[3] = (unsigned)((ga >> 32) & 0x01FFFFFFull) | 0x80000000u; // hi | type=2

    const unsigned td0 = stride_elems;                     // tensor dim0 (row len)
    const unsigned td1 = 0x00100000u;                      // tensor dim1 (ample)
    const unsigned long long s0 = stride_elems;            // dim0 stride (elems)
    v8i g1;
    g1[0] = (int)0x00020000u;                              // data_size=4B code 2
    g1[1] = (int)((td0 & 0xFFFFu) << 16);                  // td0[15:0] @ bit48
    g1[2] = (int)((td0 >> 16) | ((td1 & 0xFFFFu) << 16));  // td0[31:16] | td1[15:0]
    g1[3] = (int)((td1 >> 16) | (tile_w << 16));           // td1[31:16] | tile_dim0
    g1[4] = (int)tile_h;                                   // tile_dim1 (tile_dim2=0)
    g1[5] = (int)(s0 & 0xFFFFFFFFull);                     // dim0 stride lo
    g1[6] = (int)((s0 >> 32) & 0xFFFFull);                 // stride hi | dim1 stride=0
    g1[7] = 0;
    v4i gz = {0, 0, 0, 0};
#if defined(__clang_major__) && (__clang_major__ >= 23)
    v8i gz8 = {0, 0, 0, 0, 0, 0, 0, 0};
    __builtin_amdgcn_tensor_load_to_lds(g0, g1, gz, gz, gz8, 0);
#else
    __builtin_amdgcn_tensor_load_to_lds(g0, g1, gz, gz, 0);
#endif
}

// ---------------------------------------------------------------------------
// One 16x64 fp32 output strip of A[M,K] @ B[K,N] per wave (4 accumulators
// sharing one A fragment). Tiles staged to LDS by TDM; bf16 WMMA compute.
// Single wave (32 threads). As: 16*32 floats, Bs: 32*64 floats.
// ---------------------------------------------------------------------------
__device__ void gemm_strip16x64(const float* __restrict__ A,
                                const float* __restrict__ B,
                                int lda, int ldb, int K, int gm, int gn,
                                float* As, float* Bs, v8f acc[4]) {
    const int lane = threadIdx.x & 31;
    const int mn   = lane & 15;
    const int half = lane >> 4;
    for (int k0 = 0; k0 < K; k0 += 32) {
        // Async DMA of both tiles into LDS (2 TDM ops in flight).
        tdm_load_tile_f32(As, A + (size_t)gm * lda + k0, lda, 32u, 16u);
        tdm_load_tile_f32(Bs, B + (size_t)k0 * ldb + gn, ldb, 64u, 32u);
        __builtin_amdgcn_s_wait_tensorcnt(0);
        __syncthreads();
        v16bf af;
#pragma unroll
        for (int e = 0; e < 16; ++e)
            af[e] = (__bf16)As[mn * 32 + kmapA(e, half)];
#pragma unroll
        for (int nt = 0; nt < 4; ++nt) {
            v16bf bf;
#pragma unroll
            for (int e = 0; e < 16; ++e)
                bf[e] = (__bf16)Bs[(e + 16 * half) * 64 + nt * 16 + mn];
            acc[nt] = wmma_bf16(af, bf, acc[nt]);
        }
        __syncthreads();
    }
}

// ---------------------------------------------------------------------------
// Kernel 1: qkv = x @ Wqkv, split into head-major bf16 Q/K/V.
// Q is pre-scaled by 1/(3*rank(d)) so attention WMMAs chain without rescale.
// A 64-wide N tile lies in exactly one of q/k/v and one head (64 | 1024).
// ---------------------------------------------------------------------------
__global__ void qkv_gemm_kernel(const float* __restrict__ x,
                                const float* __restrict__ Wqkv,
                                __bf16* __restrict__ Qh,
                                __bf16* __restrict__ Kh,
                                __bf16* __restrict__ Vh) {
    __shared__ float As[16 * 32];
    __shared__ float Bs[32 * 64];
    const int gm = blockIdx.y * 16;           // token tile
    const int gn = blockIdx.x * 64;           // output-feature strip (0..3072)
    v8f acc[4] = {{}, {}, {}, {}};
    gemm_strip16x64(x, Wqkv, C_DIM, 3 * C_DIM, C_DIM, gm, gn, As, Bs, acc);

    const int lane = threadIdx.x & 31;
    const int n = lane & 15, half = lane >> 4;
    const int which = gn / C_DIM;             // 0=q 1=k 2=v
    const int h     = (gn % C_DIM) >> 6;      // head
#pragma unroll
    for (int nt = 0; nt < 4; ++nt) {
        const int d = nt * 16 + n;
#pragma unroll
        for (int r = 0; r < 8; ++r) {
            const int t = gm + r + 8 * half;
            const float v = acc[nt][r];
            const size_t idx = ((size_t)h * T_SEQ + t) * D_HEAD + d;
            if (which == 0) {
                float s = (1.0f / 3.0f) * (d < 32 ? (1.0f / 32.0f) : (1.0f / 16.0f));
                Qh[idx] = (__bf16)(v * s);
            } else if (which == 1) {
                Kh[idx] = (__bf16)v;
            } else {
                Vh[idx] = (__bf16)v;
            }
        }
    }
}

// ---------------------------------------------------------------------------
// Kernel 2: per (head, 16-row q-tile): multi-level scores -> softmax -> P@V.
// 4 waves / block. Dynamic LDS: S[16][2048] fp32 + reduction scratch (~129KB,
// fine on 320KB/WGP). Level-1/2 masks are uniform per 16x16 tile, so the
// multi-level sum is just 1-3 chained WMMAs selected by scalar branches.
// ---------------------------------------------------------------------------
__global__ void mlr_attention_kernel(const __bf16* __restrict__ Qh,
                                     const __bf16* __restrict__ Kh,
                                     const __bf16* __restrict__ Vh,
                                     float* __restrict__ y) {
    extern __shared__ float smem[];
    float* S    = smem;                 // 16 * T_SEQ scores/probs
    float* red  = S + 16 * T_SEQ;       // 8 * 16 partials
    float* rowv = red + 8 * 16;         // 16 row stats

    const int qtile = blockIdx.x;
    const int h     = blockIdx.y;
    const int qb    = qtile * 16;
    const int klen  = qb + 16;          // causal key extent
    const int tid   = threadIdx.x;
    const int lane  = tid & 31;
    const int wave  = tid >> 5;
    const int mn    = lane & 15;
    const int half  = lane >> 4;

    const __bf16* Qp = Qh + (size_t)h * T_SEQ * D_HEAD;
    const __bf16* Kp = Kh + (size_t)h * T_SEQ * D_HEAD;
    const __bf16* Vp = Vh + (size_t)h * T_SEQ * D_HEAD;

    // Per-level Q fragments (A 16x32 bf16); rank-16 levels zero-padded in K.
    v16bf aq0, aq1, aq2;
#pragma unroll
    for (int e = 0; e < 16; ++e) {
        int k = kmapA(e, half);
        const __bf16* qrow = Qp + (size_t)(qb + mn) * D_HEAD;
        aq0[e] = qrow[k];                                      // d 0..31 (rank 32)
        aq1[e] = (k < 16) ? qrow[32 + k] : (__bf16)0.0f;       // d 32..47
        aq2[e] = (k < 16) ? qrow[48 + k] : (__bf16)0.0f;       // d 48..63
    }

    // --- Phase 1: masked multi-level score tiles (chained WMMA) -------------
    for (int kt = wave; kt <= qtile; kt += 4) {
        const int kb = kt * 16;
        const bool blk1 = (qb >> 10) == (kb >> 10);   // 1024-block diagonal
        const bool blk2 = (qb >> 9)  == (kb >> 9);    // 512-block diagonal
        if (kb + 64 < T_SEQ)                           // warm L2/WGP$ ahead
            __builtin_prefetch(Kp + (size_t)(kb + 64 + mn) * D_HEAD, 0, 1);

        const __bf16* krow = Kp + (size_t)(kb + mn) * D_HEAD;  // B col = key
        v16bf b0, b1, b2;
#pragma unroll
        for (int e = 0; e < 16; ++e) {
            int k = e + 16 * half;                     // B-frag K index
            b0[e] = krow[k];
            b1[e] = (k < 16) ? krow[32 + k] : (__bf16)0.0f;
            b2[e] = (k < 16) ? krow[48 + k] : (__bf16)0.0f;
        }
        v8f s = {};
        s = wmma_bf16(aq0, b0, s);
        if (blk1) s = wmma_bf16(aq1, b1, s);
        if (blk2) s = wmma_bf16(aq2, b2, s);
#pragma unroll
        for (int r = 0; r < 8; ++r) {
            int qrow = r + 8 * half;
            float v = s[r];
            if (kt == qtile && (kb + mn) > (qb + qrow)) v = -__builtin_inff();
            S[qrow * T_SEQ + kb + mn] = v;
        }
    }
    __syncthreads();

    // --- Phase 2: row softmax over S[16][klen] ------------------------------
    {
        const int row = tid & 15, seg = tid >> 4;      // 8 segments/row
        float mx = -__builtin_inff();
        for (int j = seg; j < klen; j += 8) mx = fmaxf(mx, S[row * T_SEQ + j]);
        red[seg * 16 + row] = mx;
        __syncthreads();
        if (tid < 16) {
            float mm = red[tid];
            for (int sg = 1; sg < 8; ++sg) mm = fmaxf(mm, red[sg * 16 + tid]);
            rowv[tid] = mm;
        }
        __syncthreads();
        const float rm = rowv[row];
        float sum = 0.0f;
        for (int j = seg; j < klen; j += 8) {
            float e = __expf(S[row * T_SEQ + j] - rm);
            S[row * T_SEQ + j] = e;
            sum += e;
        }
        red[seg * 16 + row] = sum;
        __syncthreads();
        if (tid < 16) {
            float tot = red[tid];
            for (int sg = 1; sg < 8; ++sg) tot += red[sg * 16 + tid];
            rowv[tid] = 1.0f / tot;
        }
        __syncthreads();
        const float inv = rowv[row];
        for (int j = seg; j < klen; j += 8) S[row * T_SEQ + j] *= inv;
    }
    __syncthreads();

    // --- Phase 3: y[16, wave*16 : wave*16+16] = P @ V (K=32 per WMMA) -------
    {
        const int dout = wave * 16;
        v8f acc = {};
        const int nkc = (klen + 31) / 32;
        for (int kc = 0; kc < nkc; ++kc) {
            const int kb2 = kc * 32;
            v16bf ap, bv;
#pragma unroll
            for (int e = 0; e < 16; ++e) {
                int kk  = kmapA(e, half);
                int key = kb2 + kk;
                ap[e] = (key < klen) ? (__bf16)S[mn * T_SEQ + key] : (__bf16)0.0f;
                int kkb = e + 16 * half;
                bv[e] = Vp[(size_t)(kb2 + kkb) * D_HEAD + dout + mn];
            }
            acc = wmma_bf16(ap, bv, acc);
        }
#pragma unroll
        for (int r = 0; r < 8; ++r) {
            int t = qb + r + 8 * half;
            y[(size_t)t * C_DIM + h * D_HEAD + dout + mn] = acc[r];
        }
    }
}

// ---------------------------------------------------------------------------
// Kernel 3: out = y @ Wproj (fp32 out via bf16 WMMA, TDM-staged tiles).
// ---------------------------------------------------------------------------
__global__ void proj_gemm_kernel(const float* __restrict__ y,
                                 const float* __restrict__ Wproj,
                                 float* __restrict__ out) {
    __shared__ float As[16 * 32];
    __shared__ float Bs[32 * 64];
    const int gm = blockIdx.y * 16;
    const int gn = blockIdx.x * 64;
    v8f acc[4] = {{}, {}, {}, {}};
    gemm_strip16x64(y, Wproj, C_DIM, C_DIM, C_DIM, gm, gn, As, Bs, acc);

    const int lane = threadIdx.x & 31;
    const int n = lane & 15, half = lane >> 4;
#pragma unroll
    for (int nt = 0; nt < 4; ++nt)
#pragma unroll
        for (int r = 0; r < 8; ++r)
            out[(size_t)(gm + r + 8 * half) * C_DIM + gn + nt * 16 + n] =
                acc[nt][r];
}

// ---------------------------------------------------------------------------
extern "C" void kernel_launch(void* const* d_in, const int* in_sizes, int n_in,
                              void* d_out, int out_size, void* d_ws, size_t ws_size,
                              hipStream_t stream) {
    (void)in_sizes; (void)n_in; (void)out_size; (void)ws_size;
    const float* x     = (const float*)d_in[0];
    const float* Wqkv  = (const float*)d_in[1];
    const float* Wproj = (const float*)d_in[2];
    float* out = (float*)d_out;

    const size_t per = (size_t)H_NUM * T_SEQ * D_HEAD;   // 2M elems
    __bf16* Qh = (__bf16*)d_ws;                          // 4 MB
    __bf16* Kh = Qh + per;                               // 4 MB
    __bf16* Vh = Kh + per;                               // 4 MB
    float*  yb = (float*)(Vh + per);                     // 8 MB fp32 [T][C]

    dim3 g1(3 * C_DIM / 64, T_SEQ / 16);                 // 48 x 128 strips
    qkv_gemm_kernel<<<g1, 32, 0, stream>>>(x, Wqkv, Qh, Kh, Vh);

    dim3 g2(T_SEQ / 16, H_NUM);                          // 128 q-tiles x 16 heads
    size_t smem = (size_t)(16 * T_SEQ + 8 * 16 + 16) * sizeof(float);
    mlr_attention_kernel<<<g2, 128, smem, stream>>>(Qh, Kh, Vh, yb);

    dim3 g3(C_DIM / 64, T_SEQ / 16);                     // 16 x 128 strips
    proj_gemm_kernel<<<g3, 32, 0, stream>>>(yb, Wproj, out);
}
